// DIENModel_52553219834439
// MI455X (gfx1250) — matrix operand order repaired and checked by reading, not compiled, not run on hardware
//
#include <hip/hip_runtime.h>
#include <hip/hip_bf16.h>

// ---------------- problem constants (match reference) ----------------
#define Bc   2048
#define Sc   200
#define Dc   64
#define Hc   64
#define H3c  192
#define ATTc 64

typedef __attribute__((ext_vector_type(16))) __bf16 bf16x16;
typedef __attribute__((ext_vector_type(8)))  __bf16 bf16x8;
typedef __attribute__((ext_vector_type(8)))  float  floatx8;

// ---------------- WMMA helpers (CDNA5 ISA 7.12.2 layouts) ----------------
__device__ inline bf16x16 cat8(bf16x8 lo, bf16x8 hi) {
  return __builtin_shufflevector(lo, hi, 0, 1, 2, 3, 4, 5, 6, 7,
                                 8, 9, 10, 11, 12, 13, 14, 15);
}

// 16-bit A-matrix 16x32 (MxK): lane l -> M = l&15; element e (v=e>>1,pos=e&1):
// K = (v>>2)*16 + (l>>4)*8 + (v&3)*2 + pos  ==> e=0..7: K=half8+e (group 0),
// e=8..15: K=16+half8+(e-8) (group 1). Both runs are 16B-aligned in LDS.
__device__ inline bf16x16 load_a_frag(const __bf16* A, int lda, int k0, int lane) {
  const int m = lane & 15;
  const int half8 = (lane >> 4) * 8;
  const __bf16* base = A + m * lda + k0 + half8;
  const bf16x8 lo = *(const bf16x8*)(base);       // K = half8 .. half8+7
  const bf16x8 hi = *(const bf16x8*)(base + 16);  // K = 16+half8 .. 16+half8+7
  return cat8(lo, hi);
}

// B-matrix 32x16 (KxN), B[k][n] = W[n][k] for row-major weight W[N][ldw].
// lane l -> N = l&15; element e -> K = (l>>4)*8 + e (e<8), 16+(l>>4)*8+(e-8) (e>=8).
__device__ inline bf16x16 load_b_frag_wT(const __bf16* W, int ldw, int n0, int k0, int lane) {
  const int n = n0 + (lane & 15);
  const int kh = (lane >> 4) * 8;
  const __bf16* base = W + n * ldw + k0 + kh;
  const bf16x8 lo = *(const bf16x8*)(base);
  const bf16x8 hi = *(const bf16x8*)(base + 16);
  return cat8(lo, hi);
}

// f32 C/D 16x16: lane l -> N = l&15; VGPR r -> M = (l>=16 ? 8 : 0) + r.
__device__ inline void store_c_frag(float* C, int ldc, int n0, const floatx8& c, int lane) {
  const int n = n0 + (lane & 15);
  const int mb = (lane >> 4) * 8;
#pragma unroll
  for (int r = 0; r < 8; ++r) C[(mb + r) * ldc + n] = c[r];
}

__device__ inline floatx8 wmma_bf16(bf16x16 a, bf16x16 b, floatx8 c) {
  return __builtin_amdgcn_wmma_f32_16x16x32_bf16(false, a, false, b, (short)0, c, false, false);
}

__device__ inline float sigmoidf(float x) { return 1.0f / (1.0f + __expf(-x)); }

// ============================================================================
// Kernel 1: interest-extraction GRU scan. 1 WG = 16 batch rows, 200 steps.
// Per step: GI = x@Wih^T (16x192), GH = h@Whh^T (16x192) via bf16 WMMA.
// ============================================================================
__global__ __launch_bounds__(128) void dien_gru_kernel(
    const float* __restrict__ behavior, const int* __restrict__ lengths,
    const float* __restrict__ Wih, const float* __restrict__ Whh,
    const float* __restrict__ bih, const float* __restrict__ bhh,
    __bf16* __restrict__ hiddenb) {
  __shared__ __align__(16) __bf16 sWih[H3c * Dc];
  __shared__ __align__(16) __bf16 sWhh[H3c * Hc];
  __shared__ float  sBih[H3c], sBhh[H3c];
  __shared__ __align__(16) __bf16 sX[16 * Dc];
  __shared__ __align__(16) __bf16 sHb[16 * Hc];
  __shared__ float  sHf[16 * Hc];
  __shared__ float  sGI[16 * H3c];
  __shared__ float  sGH[16 * H3c];
  __shared__ int    sLen[16];

  const int tid = threadIdx.x;
  const int lane = tid & 31;
  const int wave = tid >> 5;
  const int b0 = blockIdx.x * 16;

  for (int i = tid; i < H3c * Dc; i += 128) sWih[i] = (__bf16)Wih[i];
  for (int i = tid; i < H3c * Hc; i += 128) sWhh[i] = (__bf16)Whh[i];
  for (int i = tid; i < H3c; i += 128) { sBih[i] = bih[i]; sBhh[i] = bhh[i]; }
  if (tid < 16) sLen[tid] = lengths[b0 + tid];
  for (int i = tid; i < 16 * Hc; i += 128) { sHf[i] = 0.0f; sHb[i] = (__bf16)0.0f; }
  __syncthreads();

  for (int t = 0; t < Sc; ++t) {
    for (int i = tid; i < 16 * Dc; i += 128) {
      const int m = i >> 6, d = i & 63;
      sX[i] = (__bf16)behavior[((long)(b0 + m) * Sc + t) * Dc + d];
    }
    // one global_prefetch_b8 per thread covers next step's 4KB tile at 32B pitch
    if (t + 1 < Sc) {
      const int pm = tid >> 3, pd = (tid & 7) * 8;
      __builtin_prefetch(&behavior[((long)(b0 + pm) * Sc + (t + 1)) * Dc + pd], 0, 0);
    }
    __syncthreads();

    const bf16x16 ax0 = load_a_frag(sX, Dc, 0, lane);
    const bf16x16 ax1 = load_a_frag(sX, Dc, 32, lane);
    const bf16x16 ah0 = load_a_frag(sHb, Hc, 0, lane);
    const bf16x16 ah1 = load_a_frag(sHb, Hc, 32, lane);

#pragma unroll
    for (int j = 0; j < 3; ++j) {
      const int n0 = wave * 48 + j * 16;
      floatx8 ci = {};
      ci = wmma_bf16(ax0, load_b_frag_wT(sWih, Dc, n0, 0, lane), ci);
      ci = wmma_bf16(ax1, load_b_frag_wT(sWih, Dc, n0, 32, lane), ci);
      store_c_frag(sGI, H3c, n0, ci, lane);
      floatx8 ch = {};
      ch = wmma_bf16(ah0, load_b_frag_wT(sWhh, Hc, n0, 0, lane), ch);
      ch = wmma_bf16(ah1, load_b_frag_wT(sWhh, Hc, n0, 32, lane), ch);
      store_c_frag(sGH, H3c, n0, ch, lane);
    }
    __syncthreads();

    for (int i = tid; i < 16 * Hc; i += 128) {
      const int m = i >> 6, c = i & 63;
      const float r = sigmoidf(sGI[m * H3c + c] + sBih[c] + sGH[m * H3c + c] + sBhh[c]);
      const float z = sigmoidf(sGI[m * H3c + 64 + c] + sBih[64 + c] +
                               sGH[m * H3c + 64 + c] + sBhh[64 + c]);
      const float inn = sGI[m * H3c + 128 + c] + sBih[128 + c];
      const float hn  = sGH[m * H3c + 128 + c] + sBhh[128 + c];
      const float n = tanhf(inn + r * hn);
      const float hp = sHf[i];
      float hnew = (1.0f - z) * n + z * hp;
      const bool valid = (t < sLen[m]);
      hnew = valid ? hnew : hp;
      hiddenb[((long)(b0 + m) * Sc + t) * Hc + c] = (__bf16)(valid ? hnew : 0.0f);
      sHf[i] = hnew;
      sHb[i] = (__bf16)hnew;
    }
    __syncthreads();
  }
}

// ============================================================================
// Kernel 2: attention MLP: relu([h|tgt] @ A1^T + b1) . A2, masked scores.
// 1 WG = 16 (b,s) rows; 4 waves cover N=64 with K=128 (4 WMMAs each).
// ============================================================================
__global__ __launch_bounds__(128) void dien_att_kernel(
    const __bf16* __restrict__ hiddenb, const float* __restrict__ target,
    const int* __restrict__ lengths,
    const float* __restrict__ A1, const float* __restrict__ b1,
    const float* __restrict__ A2, float* __restrict__ scores) {
  __shared__ __align__(16) __bf16 sA1[ATTc * 128];
  __shared__ float  sB1[ATTc], sA2[ATTc];
  __shared__ __align__(16) __bf16 sComb[16 * 128];
  __shared__ float  sMid[16 * ATTc];

  const int tid = threadIdx.x;
  const int lane = tid & 31;
  const int wave = tid >> 5;
  const long r0 = (long)blockIdx.x * 16;

  for (int i = tid; i < ATTc * 128; i += 128) sA1[i] = (__bf16)A1[i];
  for (int i = tid; i < ATTc; i += 128) { sB1[i] = b1[i]; sA2[i] = A2[i]; }
  for (int i = tid; i < 16 * 128; i += 128) {
    const int m = i >> 7, c = i & 127;
    const long g = r0 + m;
    const int b = (int)(g / Sc), s = (int)(g % Sc);
    const float v = (c < Hc) ? (float)hiddenb[((long)b * Sc + s) * Hc + c]
                             : target[(long)b * Dc + (c - Hc)];
    sComb[i] = (__bf16)v;
  }
  __syncthreads();

  const int n0 = wave * 16;
  floatx8 acc = {};
#pragma unroll
  for (int kk = 0; kk < 4; ++kk)
    acc = wmma_bf16(load_a_frag(sComb, 128, kk * 32, lane),
                    load_b_frag_wT(sA1, 128, n0, kk * 32, lane), acc);
  {
    const int n = n0 + (lane & 15);
    const int mb = (lane >> 4) * 8;
#pragma unroll
    for (int r = 0; r < 8; ++r) {
      const float v = acc[r] + sB1[n];
      sMid[(mb + r) * ATTc + n] = v > 0.0f ? v : 0.0f;
    }
  }
  __syncthreads();

  if (tid < 16) {
    const long g = r0 + tid;
    const int b = (int)(g / Sc), s = (int)(g % Sc);
    float dot = 0.0f;
    for (int n = 0; n < ATTc; ++n) dot += sMid[tid * ATTc + n] * sA2[n];
    scores[(long)b * Sc + s] = (s < lengths[b]) ? dot : -1e9f;
  }
}

// ============================================================================
// Kernel 3: masked softmax over S per batch row (S=200 <= 256 threads).
// ============================================================================
__global__ __launch_bounds__(256) void dien_softmax_kernel(
    const float* __restrict__ scores, float* __restrict__ attw) {
  __shared__ float red[256];
  const int tid = threadIdx.x;
  const int b = blockIdx.x;
  const float v = (tid < Sc) ? scores[(long)b * Sc + tid] : -3.0e38f;
  red[tid] = v; __syncthreads();
  for (int off = 128; off > 0; off >>= 1) {
    if (tid < off) red[tid] = fmaxf(red[tid], red[tid + off]);
    __syncthreads();
  }
  const float mx = red[0]; __syncthreads();
  const float e = (tid < Sc) ? __expf(v - mx) : 0.0f;
  red[tid] = e; __syncthreads();
  for (int off = 128; off > 0; off >>= 1) {
    if (tid < off) red[tid] += red[tid + off];
    __syncthreads();
  }
  const float sum = red[0];
  if (tid < Sc) attw[(long)b * Sc + tid] = e / sum;
}

// ============================================================================
// Kernel 4: attentional GRU scan. 1 WG = 16 batch rows, 200 steps.
// Gate matmuls split into x-part (K 0..63) and h-part (K 64..127); the n-gate
// uses r*h for its h-part as in the reference.
// ============================================================================
__global__ __launch_bounds__(128) void dien_agru_kernel(
    const __bf16* __restrict__ hiddenb, const float* __restrict__ attw,
    const float* __restrict__ Wr, const float* __restrict__ br,
    const float* __restrict__ Wz, const float* __restrict__ bz,
    const float* __restrict__ Wn, const float* __restrict__ bn,
    float* __restrict__ out) {
  __shared__ __align__(16) __bf16 sWr[Hc * 128];
  __shared__ __align__(16) __bf16 sWz[Hc * 128];
  __shared__ __align__(16) __bf16 sWn[Hc * 128];
  __shared__ float  sbr[Hc], sbz[Hc], sbn[Hc];
  __shared__ __align__(16) __bf16 sX[16 * Hc];
  __shared__ __align__(16) __bf16 sHb[16 * Hc];
  __shared__ __align__(16) __bf16 sRH[16 * Hc];
  __shared__ float  sHf[16 * Hc], sRg[16 * Hc], sZg[16 * Hc];
  __shared__ float  sA[16];

  const int tid = threadIdx.x;
  const int lane = tid & 31;
  const int wave = tid >> 5;
  const int b0 = blockIdx.x * 16;

  for (int i = tid; i < Hc * 128; i += 128) {
    sWr[i] = (__bf16)Wr[i]; sWz[i] = (__bf16)Wz[i]; sWn[i] = (__bf16)Wn[i];
  }
  for (int i = tid; i < Hc; i += 128) { sbr[i] = br[i]; sbz[i] = bz[i]; sbn[i] = bn[i]; }
  for (int i = tid; i < 16 * Hc; i += 128) { sHf[i] = 0.0f; sHb[i] = (__bf16)0.0f; }
  __syncthreads();

  const int n0 = wave * 16;
  for (int t = 0; t < Sc; ++t) {
    for (int i = tid; i < 16 * Hc; i += 128) {
      const int m = i >> 6, c = i & 63;
      sX[i] = hiddenb[((long)(b0 + m) * Sc + t) * Hc + c];
    }
    if (tid < 16) sA[tid] = attw[(long)(b0 + tid) * Sc + t];
    __syncthreads();

    const bf16x16 ax0 = load_a_frag(sX, Hc, 0, lane);
    const bf16x16 ax1 = load_a_frag(sX, Hc, 32, lane);
    const bf16x16 ah0 = load_a_frag(sHb, Hc, 0, lane);
    const bf16x16 ah1 = load_a_frag(sHb, Hc, 32, lane);

    floatx8 cr = {}, cz = {};
    cr = wmma_bf16(ax0, load_b_frag_wT(sWr, 128, n0, 0, lane), cr);
    cr = wmma_bf16(ax1, load_b_frag_wT(sWr, 128, n0, 32, lane), cr);
    cr = wmma_bf16(ah0, load_b_frag_wT(sWr, 128, n0, 64, lane), cr);
    cr = wmma_bf16(ah1, load_b_frag_wT(sWr, 128, n0, 96, lane), cr);
    store_c_frag(sRg, Hc, n0, cr, lane);
    cz = wmma_bf16(ax0, load_b_frag_wT(sWz, 128, n0, 0, lane), cz);
    cz = wmma_bf16(ax1, load_b_frag_wT(sWz, 128, n0, 32, lane), cz);
    cz = wmma_bf16(ah0, load_b_frag_wT(sWz, 128, n0, 64, lane), cz);
    cz = wmma_bf16(ah1, load_b_frag_wT(sWz, 128, n0, 96, lane), cz);
    store_c_frag(sZg, Hc, n0, cz, lane);
    __syncthreads();

    for (int i = tid; i < 16 * Hc; i += 128) {
      const int m = i >> 6, c = i & 63;
      const float r = sigmoidf(sRg[i] + sbr[c]);
      const float z = sigmoidf(sZg[i] + sbz[c]) * sA[m];
      sZg[i] = z;
      sRH[i] = (__bf16)(r * sHf[i]);
    }
    __syncthreads();

    const bf16x16 arh0 = load_a_frag(sRH, Hc, 0, lane);
    const bf16x16 arh1 = load_a_frag(sRH, Hc, 32, lane);
    floatx8 cn = {};
    cn = wmma_bf16(ax0, load_b_frag_wT(sWn, 128, n0, 0, lane), cn);
    cn = wmma_bf16(ax1, load_b_frag_wT(sWn, 128, n0, 32, lane), cn);
    cn = wmma_bf16(arh0, load_b_frag_wT(sWn, 128, n0, 64, lane), cn);
    cn = wmma_bf16(arh1, load_b_frag_wT(sWn, 128, n0, 96, lane), cn);
    store_c_frag(sRg, Hc, n0, cn, lane);  // reuse sRg as n-gate staging
    __syncthreads();

    for (int i = tid; i < 16 * Hc; i += 128) {
      const int c = i & 63;
      const float n = tanhf(sRg[i] + sbn[c]);
      const float z = sZg[i];
      const float h = sHf[i];
      const float hnew = (1.0f - z) * h + z * n;
      sHf[i] = hnew;
      sHb[i] = (__bf16)hnew;
    }
    __syncthreads();
  }

  for (int i = tid; i < 16 * Hc; i += 128) {
    const int m = i >> 6, c = i & 63;
    out[(long)(b0 + m) * Hc + c] = sHf[i];
  }
}

// ============================================================================
extern "C" void kernel_launch(void* const* d_in, const int* in_sizes, int n_in,
                              void* d_out, int out_size, void* d_ws, size_t ws_size,
                              hipStream_t stream) {
  const float* behavior = (const float*)d_in[0];
  const float* target   = (const float*)d_in[1];
  const int*   lengths  = (const int*)d_in[2];
  const float* Wih = (const float*)d_in[3];
  const float* Whh = (const float*)d_in[4];
  const float* bih = (const float*)d_in[5];
  const float* bhh = (const float*)d_in[6];
  const float* A1  = (const float*)d_in[7];
  const float* b1  = (const float*)d_in[8];
  const float* A2  = (const float*)d_in[9];
  const float* Wr  = (const float*)d_in[10];
  const float* br  = (const float*)d_in[11];
  const float* Wz  = (const float*)d_in[12];
  const float* bz  = (const float*)d_in[13];
  const float* Wn  = (const float*)d_in[14];
  const float* bn  = (const float*)d_in[15];

  char* ws = (char*)d_ws;
  __bf16* hiddenb = (__bf16*)ws;                                  // B*S*H bf16 (~52 MB)
  float*  scores  = (float*)(ws + (size_t)Bc * Sc * Hc * 2);      // B*S f32
  float*  attw    = scores + (size_t)Bc * Sc;                     // B*S f32
  float*  outp    = (float*)d_out;

  dien_gru_kernel<<<Bc / 16, 128, 0, stream>>>(behavior, lengths, Wih, Whh, bih, bhh, hiddenb);
  dien_att_kernel<<<(Bc * Sc) / 16, 128, 0, stream>>>(hiddenb, target, lengths, A1, b1, A2, scores);
  dien_softmax_kernel<<<Bc, 256, 0, stream>>>(scores, attw);
  dien_agru_kernel<<<Bc / 16, 128, 0, stream>>>(hiddenb, attw, Wr, br, Wz, bz, Wn, bn, outp);
}
